// RoPEBlock_78280073937012
// MI455X (gfx1250) — compile-verified, matrix operand order checked by reading, not run on hardware
//
#include <hip/hip_runtime.h>
#include <hip/hip_bf16.h>
#include <math.h>

// ---------------------------------------------------------------------------
// Types for CDNA5 WMMA (wave32, 16x16x32 bf16 -> f32)
// ---------------------------------------------------------------------------
typedef __attribute__((ext_vector_type(16))) __bf16 v16bf;
typedef __attribute__((ext_vector_type(8)))  __bf16 v8bf;
typedef __attribute__((ext_vector_type(8)))  float  v8f;

#define BATCH 8
#define SEQ   1024
#define DIM   1024
#define HEADS 16
#define HDIM  64

__device__ __forceinline__ v16bf load_frag16(const __bf16* p) {
  // A/B operand layout for 16-bit WMMA: 8 contiguous bf16 at +0 (K=kbase..+7)
  // and 8 contiguous bf16 at +16 (K=16+kbase..+7). Two b128 loads.
  v8bf lo = *(const v8bf*)(p);
  v8bf hi = *(const v8bf*)(p + 16);
  return __builtin_shufflevector(lo, hi, 0,1,2,3,4,5,6,7,8,9,10,11,12,13,14,15);
}

__device__ __forceinline__ v8f wmma_bf16(v16bf a, v16bf b, v8f c) {
  return __builtin_amdgcn_wmma_f32_16x16x32_bf16(false, a, false, b, (short)0, c,
                                                 false, false);
}

// ---------------------------------------------------------------------------
// CDNA5 async global->LDS copy (ASYNCcnt tracked, no VGPR data movement).
// LDS byte address (wave-relative) = low 32 bits of generic shared pointer.
// ---------------------------------------------------------------------------
__device__ __forceinline__ unsigned lds_addr32(const void* p) {
  return (unsigned)(unsigned long long)p;
}

__device__ __forceinline__ void async_copy_b128(unsigned lds_off,
                                                const void* gptr) {
  asm volatile("global_load_async_to_lds_b128 %0, %1, off"
               :
               : "v"(lds_off), "v"((unsigned long long)gptr)
               : "memory");
}

__device__ __forceinline__ void wait_async0() {
  asm volatile("s_wait_asynccnt 0x0" ::: "memory");
}

// ---------------------------------------------------------------------------
// Weight prep: f32 [K,N] -> bf16 transposed [N,K] (so GEMM B-tiles are
// K-contiguous and can be staged with async b128 copies, no conversion).
// ---------------------------------------------------------------------------
__global__ void wt_transpose_bf16_kernel(const float* __restrict__ src,
                                         __bf16* __restrict__ dst, int K,
                                         int N) {
  const int idx = blockIdx.x * blockDim.x + threadIdx.x;
  if (idx >= K * N) return;
  const int n = idx / K;
  const int k = idx - n * K;
  dst[(size_t)n * K + k] = (__bf16)src[(size_t)k * N + n];
}

// ---------------------------------------------------------------------------
// RMSNorm (reduction over axes (1,2): whole (S,D) plane per batch sample)
// ---------------------------------------------------------------------------
__global__ void rms_partial_kernel(const float* __restrict__ x,
                                   float* __restrict__ sums, int nd) {
  __shared__ float red[256];
  const int b = blockIdx.y;
  const float* p = x + (size_t)b * nd;
  float acc = 0.f;
  for (int i = blockIdx.x * blockDim.x + threadIdx.x; i < nd;
       i += gridDim.x * blockDim.x) {
    float v = p[i];
    acc += v * v;
  }
  red[threadIdx.x] = acc;
  __syncthreads();
  for (int s = 128; s > 0; s >>= 1) {
    if (threadIdx.x < (unsigned)s) red[threadIdx.x] += red[threadIdx.x + s];
    __syncthreads();
  }
  if (threadIdx.x == 0) atomicAdd(&sums[b], red[0]);
}

__global__ void rms_apply_kernel(const float* __restrict__ x,
                                 const float* __restrict__ scale,
                                 const float* __restrict__ sums,
                                 float* __restrict__ out, int nd) {
  const int b = blockIdx.y;
  const size_t base = (size_t)b * nd;
  const float inv = rsqrtf(sums[b] * (1.0f / (float)nd));
  for (int i = blockIdx.x * blockDim.x + threadIdx.x; i < nd;
       i += gridDim.x * blockDim.x)
    out[base + i] = scale[i] * x[base + i] * inv;
}

// ---------------------------------------------------------------------------
// GEMM: C[M,N] = A[M,K](f32) * W (bf16, stored transposed [N,K])
//        (+bias[N]) (+resid[M,N])
// 128x128x32 tiles, 256 threads = 8 waves (4x2), 32x64 per wave.
// Double-buffered LDS; B tiles staged with async global->LDS copies that
// overlap the WMMAs of the current tile; A staged with batched loads + cvt.
// ---------------------------------------------------------------------------
#define GBM 128
#define GBN 128
#define GBK 32
#define GLD 40  // padded LDS row stride in bf16 elements (80B, 16B-multiple)

__global__ __launch_bounds__(256) void gemm_bf16_kernel(
    const float* __restrict__ A, const __bf16* __restrict__ BT,
    const float* __restrict__ bias, const float* __restrict__ resid,
    float* __restrict__ C, int M, int N, int K) {
  __shared__ alignas(16) __bf16 Asb[2][GBM * GLD];
  __shared__ alignas(16) __bf16 Btb[2][GBN * GLD];  // [n][k]

  const int tid  = threadIdx.x;
  const int wave = tid >> 5;
  const int lane = tid & 31;
  const int wm = wave >> 1;          // 0..3 (M direction)
  const int wn = wave & 1;           // 0..1 (N direction)
  const int hgrp = lane >> 4;        // which 16-lane half
  const int lrow = lane & 15;
  const int kbase = hgrp * 8;

  const int m0 = blockIdx.y * GBM;
  const int n0 = blockIdx.x * GBN;

  v8f acc[2][4];
#pragma unroll
  for (int tm = 0; tm < 2; ++tm)
#pragma unroll
    for (int tn = 0; tn < 4; ++tn) acc[tm][tn] = (v8f){};

  const int arow = tid >> 3;         // 0..31
  const int acol = (tid & 7) * 4;    // 0..28
  const int bn   = tid >> 1;         // 0..127 (row of B^T)
  const int bks  = (tid & 1) * 16;   // 0 or 16 (k sub-chunk)

  auto stage = [&](int buf, int k0) {
    // B tile: pure bf16 copy -> async global->LDS (2 x b128 per thread)
    const __bf16* g = BT + (size_t)(n0 + bn) * K + k0 + bks;
    const unsigned l = lds_addr32(&Btb[buf][bn * GLD + bks]);
    async_copy_b128(l, g);
    async_copy_b128(l + 16, g + 8);
    // A tile: batch all 4 loads first (MLP), then convert + store
    float4 va[4];
#pragma unroll
    for (int p = 0; p < 4; ++p)
      va[p] = *(const float4*)(A + (size_t)(m0 + arow + 32 * p) * K + k0 + acol);
    if (k0 + GBK < K)  // hint following A tile (global_prefetch_b8)
      __builtin_prefetch((const void*)(A + (size_t)(m0 + arow) * K + k0 + GBK), 0, 0);
#pragma unroll
    for (int p = 0; p < 4; ++p) {
      __bf16* d = &Asb[buf][(arow + 32 * p) * GLD + acol];
      d[0] = (__bf16)va[p].x; d[1] = (__bf16)va[p].y;
      d[2] = (__bf16)va[p].z; d[3] = (__bf16)va[p].w;
    }
  };

  stage(0, 0);
  const int nk = K / GBK;
  for (int it = 0; it < nk; ++it) {
    wait_async0();      // own async copies into buf[it&1] done
    __syncthreads();    // everyone's staging of buf[it&1] visible
    if (it + 1 < nk) stage((it + 1) & 1, (it + 1) * GBK);  // overlap next tile

    const __bf16* As = &Asb[it & 1][0];
    const __bf16* Bs = &Btb[it & 1][0];
    v16bf af[2], bfr[4];
#pragma unroll
    for (int tm = 0; tm < 2; ++tm)
      af[tm] = load_frag16(&As[(wm * 32 + tm * 16 + lrow) * GLD + kbase]);
#pragma unroll
    for (int tn = 0; tn < 4; ++tn)
      bfr[tn] = load_frag16(&Bs[(wn * 64 + tn * 16 + lrow) * GLD + kbase]);
#pragma unroll
    for (int tm = 0; tm < 2; ++tm)
#pragma unroll
      for (int tn = 0; tn < 4; ++tn)
        acc[tm][tn] = wmma_bf16(af[tm], bfr[tn], acc[tm][tn]);
  }

  // epilogue: C-layout = VGPR j, lanes 0-15 -> row j, lanes 16-31 -> row j+8
#pragma unroll
  for (int tm = 0; tm < 2; ++tm)
#pragma unroll
    for (int tn = 0; tn < 4; ++tn)
#pragma unroll
      for (int j = 0; j < 8; ++j) {
        const int row = m0 + wm * 32 + tm * 16 + j + 8 * hgrp;
        const int col = n0 + wn * 64 + tn * 16 + lrow;
        float v = acc[tm][tn][j];
        if (bias)  v += bias[col];
        const size_t idx = (size_t)row * N + col;
        if (resid) v += resid[idx];
        C[idx] = v;
      }
}

// ---------------------------------------------------------------------------
// RoPE + head split + bf16 conversion.
// q_rot[2i]   = q[2i]*c - q[2i+1]*s ; q_rot[2i+1] = q[2i]*s + q[2i+1]*c
// c = R[m,2i,2i], s = R[m,2i+1,2i]  (read from the dense R input).
// Output layout: [B, H, S, HD] bf16.
// ---------------------------------------------------------------------------
__global__ void rope_split_kernel(const float* __restrict__ qf,
                                  const float* __restrict__ kf,
                                  const float* __restrict__ vf,
                                  const float* __restrict__ R,
                                  __bf16* __restrict__ Qr,
                                  __bf16* __restrict__ Kr,
                                  __bf16* __restrict__ Vb) {
  const int idx = blockIdx.x * blockDim.x + threadIdx.x;  // pair index
  const int npairs = BATCH * SEQ * HEADS * (HDIM / 2);
  if (idx >= npairs) return;
  const int i2 = idx & 31;                 // pair within head
  const int h  = (idx >> 5) & (HEADS - 1);
  const int s  = (idx >> 9) & (SEQ - 1);
  const int b  = idx >> 19;
  const int d0 = 2 * i2;

  const size_t src = ((size_t)b * SEQ + s) * DIM + h * HDIM + d0;
  const size_t dst = (((size_t)b * HEADS + h) * SEQ + s) * HDIM + d0;
  const float c  = R[((size_t)s * HDIM + d0) * HDIM + d0];
  const float sn = R[((size_t)s * HDIM + d0 + 1) * HDIM + d0];

  const float q0 = qf[src], q1 = qf[src + 1];
  const float k0 = kf[src], k1 = kf[src + 1];
  Qr[dst]     = (__bf16)(q0 * c - q1 * sn);
  Qr[dst + 1] = (__bf16)(q0 * sn + q1 * c);
  Kr[dst]     = (__bf16)(k0 * c - k1 * sn);
  Kr[dst + 1] = (__bf16)(k0 * sn + k1 * c);
  Vb[dst]     = (__bf16)vf[src];
  Vb[dst + 1] = (__bf16)vf[src + 1];
}

// ---------------------------------------------------------------------------
// Flash attention: one block = (b, h, 128-query tile). 8 waves; each wave owns
// a 16-row query strip and streams 64-key tiles. K tiles staged with async
// global->LDS copies; V transposed manually. WMMA for QK^T and PV, online
// softmax with 16-lane shuffle reductions. Writes x2 = x1 + attn.
// ---------------------------------------------------------------------------
#define ALD 72  // padded LDS stride (144B, 16B-multiple)

__global__ __launch_bounds__(256) void attention_kernel(
    const __bf16* __restrict__ Qr, const __bf16* __restrict__ Kr,
    const __bf16* __restrict__ Vb, const float* __restrict__ x1,
    float* __restrict__ x2) {
  __shared__ alignas(16) __bf16 Kt[64 * ALD];         // [key][hd]
  __shared__ alignas(16) __bf16 Vt[64 * ALD];         // [hd][key] (transposed)
  __shared__ alignas(16) __bf16 Pl[8 * 16 * ALD];     // per-wave P round-trip

  const int tid  = threadIdx.x;
  const int wave = tid >> 5;
  const int lane = tid & 31;
  const int hgrp = lane >> 4;
  const int lrow = lane & 15;
  const int kbase = hgrp * 8;

  const int qt = blockIdx.x;               // query tile (128 rows)
  const int h  = blockIdx.y;
  const int b  = blockIdx.z;
  const size_t bh = ((size_t)b * HEADS + h) * SEQ;

  const int q0 = qt * 128 + wave * 16;

  // Q fragments straight from global (row-major [.,HD], b128 loads)
  v16bf aq[2];
  {
    const __bf16* qp = Qr + (bh + q0 + lrow) * HDIM;
    aq[0] = load_frag16(qp + kbase);
    aq[1] = load_frag16(qp + 32 + kbase);
  }

  v8f o[4];
#pragma unroll
  for (int tn = 0; tn < 4; ++tn) o[tn] = (v8f){};
  float rmax[8], rsum[8];
#pragma unroll
  for (int r = 0; r < 8; ++r) { rmax[r] = -1e30f; rsum[r] = 0.f; }

  const int crow = tid >> 3;        // 0..31
  const int ccol = (tid & 7) * 8;   // 0..56

  for (int j = 0; j < SEQ / 64; ++j) {
    __syncthreads();
    // K tile: async global->LDS (row-major copy, 2 x b128 per thread)
#pragma unroll
    for (int p = 0; p < 2; ++p) {
      const int r = crow + 32 * p;
      async_copy_b128(lds_addr32(&Kt[r * ALD + ccol]),
                      Kr + (bh + j * 64 + r) * HDIM + ccol);
    }
    // V tile: batch both loads, then scatter-transpose into LDS
    v8bf vv[2];
#pragma unroll
    for (int p = 0; p < 2; ++p)
      vv[p] = *(const v8bf*)(Vb + (bh + j * 64 + crow + 32 * p) * HDIM + ccol);
#pragma unroll
    for (int p = 0; p < 2; ++p)
#pragma unroll
      for (int i = 0; i < 8; ++i)
        Vt[(ccol + i) * ALD + crow + 32 * p] = vv[p][i];
    wait_async0();
    __syncthreads();

    // scores S = Q * K^T   (B operand: column n = key row, contiguous in LDS)
    v8f sc[4];
#pragma unroll
    for (int tn = 0; tn < 4; ++tn) sc[tn] = (v8f){};
#pragma unroll
    for (int tn = 0; tn < 4; ++tn) {
      const __bf16* kp = &Kt[(tn * 16 + lrow) * ALD];
      v16bf bk0 = load_frag16(kp + kbase);
      v16bf bk1 = load_frag16(kp + 32 + kbase);
      sc[tn] = wmma_bf16(aq[0], bk0, sc[tn]);
      sc[tn] = wmma_bf16(aq[1], bk1, sc[tn]);
    }

    // online softmax (scale 1/sqrt(64) = 0.125)
#pragma unroll
    for (int tn = 0; tn < 4; ++tn)
#pragma unroll
      for (int r = 0; r < 8; ++r) sc[tn][r] *= 0.125f;

#pragma unroll
    for (int r = 0; r < 8; ++r) {
      float mx = fmaxf(fmaxf(sc[0][r], sc[1][r]), fmaxf(sc[2][r], sc[3][r]));
#pragma unroll
      for (int off = 8; off >= 1; off >>= 1)
        mx = fmaxf(mx, __shfl_xor(mx, off, 32));   // stays in 16-lane half
      const float nm = fmaxf(rmax[r], mx);
      const float corr = __expf(rmax[r] - nm);
      rmax[r] = nm;
      float ps = 0.f;
#pragma unroll
      for (int tn = 0; tn < 4; ++tn) {
        const float p = __expf(sc[tn][r] - nm);
        sc[tn][r] = p;
        ps += p;
      }
#pragma unroll
      for (int off = 8; off >= 1; off >>= 1) ps += __shfl_xor(ps, off, 32);
      rsum[r] = rsum[r] * corr + ps;
#pragma unroll
      for (int tn = 0; tn < 4; ++tn) o[tn][r] *= corr;
    }

    // P: C-layout regs -> A-layout via per-wave LDS round trip
    __bf16* pw = &Pl[wave * 16 * ALD];
#pragma unroll
    for (int tn = 0; tn < 4; ++tn)
#pragma unroll
      for (int r = 0; r < 8; ++r)
        pw[(r + 8 * hgrp) * ALD + tn * 16 + lrow] = (__bf16)sc[tn][r];
    __syncthreads();

    v16bf ap0 = load_frag16(pw + lrow * ALD + kbase);
    v16bf ap1 = load_frag16(pw + lrow * ALD + 32 + kbase);
#pragma unroll
    for (int tn = 0; tn < 4; ++tn) {
      const __bf16* vp = &Vt[(tn * 16 + lrow) * ALD];
      v16bf bv0 = load_frag16(vp + kbase);
      v16bf bv1 = load_frag16(vp + 32 + kbase);
      o[tn] = wmma_bf16(ap0, bv0, o[tn]);
      o[tn] = wmma_bf16(ap1, bv1, o[tn]);
    }
  }

  // finalize: x2 = x1 + O / rowsum
#pragma unroll
  for (int tn = 0; tn < 4; ++tn)
#pragma unroll
    for (int r = 0; r < 8; ++r) {
      const int qrow = q0 + r + 8 * hgrp;
      const int dcol = h * HDIM + tn * 16 + lrow;
      const size_t idx = ((size_t)b * SEQ + qrow) * DIM + dcol;
      x2[idx] = x1[idx] + o[tn][r] / rsum[r];
    }
}

// ---------------------------------------------------------------------------
// SwiGLU gate: sact = (g * sigmoid(beta*g)) * l
// ---------------------------------------------------------------------------
__global__ void swiglu_kernel(const float* __restrict__ g,
                              const float* __restrict__ l,
                              const float* __restrict__ beta,
                              float* __restrict__ out, int n) {
  const float bta = beta[0];
  for (int i = blockIdx.x * blockDim.x + threadIdx.x; i < n;
       i += gridDim.x * blockDim.x) {
    const float gv = g[i];
    const float sg = gv / (1.f + __expf(-bta * gv));
    out[i] = sg * l[i];
  }
}

// ---------------------------------------------------------------------------
// Host-side launch
// ---------------------------------------------------------------------------
extern "C" void kernel_launch(void* const* d_in, const int* in_sizes, int n_in,
                              void* d_out, int out_size, void* d_ws,
                              size_t ws_size, hipStream_t stream) {
  (void)in_sizes; (void)n_in; (void)out_size; (void)ws_size;
  const float* x     = (const float*)d_in[0];
  const float* scale = (const float*)d_in[1];
  const float* wq    = (const float*)d_in[2];
  const float* wk    = (const float*)d_in[3];
  const float* wv    = (const float*)d_in[4];
  const float* R     = (const float*)d_in[5];
  const float* w1    = (const float*)d_in[6];
  const float* b1    = (const float*)d_in[7];
  const float* wg    = (const float*)d_in[8];
  const float* bg    = (const float*)d_in[9];
  const float* wl    = (const float*)d_in[10];
  const float* bl    = (const float*)d_in[11];
  const float* beta  = (const float*)d_in[12];
  const float* w2    = (const float*)d_in[13];
  const float* b2    = (const float*)d_in[14];
  float* out = (float*)d_out;

  const int M  = BATCH * SEQ;      // 8192
  const int N  = DIM;              // 1024
  const int K  = DIM;              // 1024
  const int ND = SEQ * DIM;        // per-batch plane
  const size_t NELT = (size_t)M * N;           // 8M
  const size_t F32B = NELT * sizeof(float);    // 32 MiB
  const size_t BF6B = NELT * sizeof(__bf16);   // 16 MiB
  const size_t WBFB = (size_t)N * K * sizeof(__bf16);  // 2 MiB

  // workspace layout
  char* w = (char*)d_ws;
  float* sums0 = (float*)w;                      // 8 floats
  float* sums1 = (float*)(w + 32);               // 8 floats
  size_t off = 256;
  __bf16* wbf[7];
  for (int i = 0; i < 7; ++i) { wbf[i] = (__bf16*)(w + off); off += WBFB; }
  float* x1   = (float*)(w + off); off += F32B;
  float* bufQ = (float*)(w + off); off += F32B;  // qf, later h
  float* bufK = (float*)(w + off); off += F32B;  // kf, later g
  float* bufV = (float*)(w + off); off += F32B;  // vf, later l
  float* x2   = (float*)(w + off); off += F32B;  // later sact
  float* x3   = (float*)(w + off); off += F32B;
  __bf16* Qr  = (__bf16*)(w + off); off += BF6B;
  __bf16* Kr  = (__bf16*)(w + off); off += BF6B;
  __bf16* Vb  = (__bf16*)(w + off); off += BF6B;

  hipMemsetAsync(w, 0, 64, stream);

  // weights -> bf16, transposed to [N,K] for async K-contiguous staging
  const float* wsrc[7] = {wq, wk, wv, w1, wg, wl, w2};
  const int nconv = (N * K + 255) / 256;
  for (int i = 0; i < 7; ++i)
    wt_transpose_bf16_kernel<<<nconv, 256, 0, stream>>>(wsrc[i], wbf[i], K, N);

  const dim3 gG(N / GBN, M / GBM);   // (8, 64)
  const dim3 rG(64, BATCH);

  // x1 = rmsnorm(x)
  rms_partial_kernel<<<rG, 256, 0, stream>>>(x, sums0, ND);
  rms_apply_kernel<<<rG, 256, 0, stream>>>(x, scale, sums0, x1, ND);

  // QKV projections
  gemm_bf16_kernel<<<gG, 256, 0, stream>>>(x1, wbf[0], nullptr, nullptr, bufQ, M, N, K);
  gemm_bf16_kernel<<<gG, 256, 0, stream>>>(x1, wbf[1], nullptr, nullptr, bufK, M, N, K);
  gemm_bf16_kernel<<<gG, 256, 0, stream>>>(x1, wbf[2], nullptr, nullptr, bufV, M, N, K);

  // RoPE + head split + bf16
  const int npairs = BATCH * SEQ * HEADS * (HDIM / 2);
  rope_split_kernel<<<(npairs + 255) / 256, 256, 0, stream>>>(bufQ, bufK, bufV, R,
                                                              Qr, Kr, Vb);

  // attention + residual: x2 = x1 + attn
  attention_kernel<<<dim3(SEQ / 128, HEADS, BATCH), 256, 0, stream>>>(Qr, Kr, Vb,
                                                                      x1, x2);

  // x3 = rmsnorm(x2)
  rms_partial_kernel<<<rG, 256, 0, stream>>>(x2, sums1, ND);
  rms_apply_kernel<<<rG, 256, 0, stream>>>(x2, scale, sums1, x3, ND);

  // FFN: h = x3@w1+b1 ; g = h@wg+bg ; l = h@wl+bl
  float* hbuf = bufQ; float* gbuf = bufK; float* lbuf = bufV;
  gemm_bf16_kernel<<<gG, 256, 0, stream>>>(x3, wbf[3], b1, nullptr, hbuf, M, N, K);
  gemm_bf16_kernel<<<gG, 256, 0, stream>>>(hbuf, wbf[4], bg, nullptr, gbuf, M, N, K);
  gemm_bf16_kernel<<<gG, 256, 0, stream>>>(hbuf, wbf[5], bl, nullptr, lbuf, M, N, K);

  // sact = swish(g) * l   (reuse x2 buffer; x2 no longer needed)
  float* sact = x2;
  swiglu_kernel<<<8192, 256, 0, stream>>>(gbuf, lbuf, beta, sact, (int)NELT);

  // out = x3 + sact@w2 + b2
  gemm_bf16_kernel<<<gG, 256, 0, stream>>>(sact, wbf[6], b2, x3, out, M, N, K);
}
